// stacked_rnn_80633716015529
// MI455X (gfx1250) — matrix operand order repaired and checked by reading, not compiled, not run on hardware
//
#include <hip/hip_runtime.h>
#include <hip/hip_bf16.h>

typedef __attribute__((ext_vector_type(16))) __bf16 v16bf;
typedef __attribute__((ext_vector_type(8)))  float  v8f;

#define DEV __device__ __forceinline__

static DEV unsigned short f32_to_bf16_bits(float f) {
  unsigned int u = __float_as_uint(f);
  unsigned int r = (u + 0x7FFFu + ((u >> 16) & 1u)) >> 16;  // round-to-nearest-even
  return (unsigned short)r;
}

union FragU { uint4 q[2]; v16bf v; };

// Load one 512-short fragment tile lane-slice (32B) at byte offset `off` from a
// per-lane base pointer. After unrolling, `off` folds into the b128 immediate.
static DEV void load_frag_at(const char* __restrict__ p, int off, FragU& u) {
  u.q[0] = *reinterpret_cast<const uint4*>(p + off);
  u.q[1] = *reinterpret_cast<const uint4*>(p + off + 16);
}

// ---------------- setup kernels ----------------

// Pack W [N x K] fp32 row-major as WMMA B fragments (B[k][n] = W[n][k]).
// B 32x16 layout: lane = n_local + 16*((kk>>3)&1), element e = (kk&7) + 8*(kk>=16).
__global__ void pack_weight_b(const float* __restrict__ W, int K, int nKt,
                              unsigned short* __restrict__ out, int total) {
  int id = blockIdx.x * blockDim.x + threadIdx.x;
  if (id >= total) return;
  int e  = id & 15;
  int l  = (id >> 4) & 31;
  int t2 = id >> 9;
  int kt = t2 % nKt;
  int nt = t2 / nKt;
  int n  = nt * 16 + (l & 15);
  int kk = kt * 32 + (e & 7) + ((l >> 4) << 3) + ((e >> 3) << 4);
  float v = (kk < K) ? W[(size_t)n * K + kk] : 0.f;
  out[id] = f32_to_bf16_bits(v);
}

// Pack x [B,T,150] fp32 as WMMA A fragments per timestep, K padded to 160.
// A 16x32 layout: lane = m_local + 16*((kk>>3)&1), element e = (kk&7) + 8*(kk>=16).
__global__ void pack_x_a(const float* __restrict__ x, unsigned short* __restrict__ out,
                         int total) {
  int id = blockIdx.x * blockDim.x + threadIdx.x;
  if (id >= total) return;
  int e = id & 15;
  int l = (id >> 4) & 31;
  int r = id >> 9;
  int kt = r % 5; r /= 5;
  int mt = r & 7;
  int t  = r >> 3;
  int m    = mt * 16 + (l & 15);
  int half = l >> 4;
  int kk   = kt * 32 + (e & 7) + (half << 3) + ((e >> 3) << 4);
  float v = (kk < 150) ? x[((size_t)m * 256 + t) * 150 + kk] : 0.f;
  out[id] = f32_to_bf16_bits(v);
}

// Pack h [128 x 1024] fp32 as A fragments (initial state).
__global__ void pack_h_a(const float* __restrict__ h, unsigned short* __restrict__ out) {
  int id = blockIdx.x * blockDim.x + threadIdx.x;  // 131072 threads
  int e = id & 15;
  int l = (id >> 4) & 31;
  int r = id >> 9;
  int kt = r & 31;
  int mt = r >> 5;
  int half = l >> 4;
  int kk   = kt * 32 + (e & 7) + (half << 3) + ((e >> 3) << 4);
  int m    = mt * 16 + (l & 15);
  out[id] = f32_to_bf16_bits(h[(size_t)m * 1024 + kk]);
}

__global__ void combine_bias(const float* __restrict__ a1, const float* __restrict__ a2,
                             const float* __restrict__ b1, const float* __restrict__ b2,
                             float* __restrict__ o1, float* __restrict__ o2) {
  int i = blockIdx.x * blockDim.x + threadIdx.x;
  if (i < 4096) {
    o1[i] = a1[i] + a2[i];
    o2[i] = b1[i] + b2[i];
  }
}

// ---------------- fused LSTM layer step ----------------
// gates = A0@B0 + A1@B1 + bias; then cell update; h written back as packed A frags.
// Grid: (8 M-tiles, 32 column groups of 32 within H). 8 waves = (gate g, sub-col).
__global__ __launch_bounds__(256)
void lstm_layer_kernel(const unsigned short* __restrict__ A0, int nKt0,
                       const unsigned short* __restrict__ B0,
                       const unsigned short* __restrict__ A1,
                       const unsigned short* __restrict__ B1,
                       const float* __restrict__ bias,        // [4096], i|f|g|o
                       float* __restrict__ cbuf,              // [128*1024], in/out
                       unsigned short* __restrict__ h_packed, // next-step A frags
                       float* __restrict__ h_f32)             // optional fp32 copy
{
  __shared__ float lds[4][16][32];
  const int tid  = threadIdx.x;
  const int lane = tid & 31;
  const int wave = tid >> 5;
  const int g    = wave >> 1;        // gate 0..3 (i,f,g,o)
  const int sub  = wave & 1;         // 16-col sub-tile
  const int mt   = blockIdx.x;       // 0..7
  const int cg   = blockIdx.y;       // 0..31
  const int colInGate = cg * 32 + sub * 16;
  const int ntile     = (g * 1024 + colInGate) >> 4;
  const int laneByte  = lane * 32;   // 16 shorts per lane-slice

  v8f acc = {};

  // --- pass 0: x (or h1) contribution, nKt0 tiles of K=32 ---
  {
    const char* pa = (const char*)A0 + (size_t)mt * nKt0 * 1024 + laneByte;
    const char* pb = (const char*)B0 + (size_t)ntile * nKt0 * 1024 + laneByte;
    FragU a0, b0, a1, b1;
    load_frag_at(pa, 0, a0);
    load_frag_at(pb, 0, b0);
    for (int kt = 0; kt < nKt0 - 1; ++kt) {
      load_frag_at(pa, (kt + 1) * 1024, a1);
      load_frag_at(pb, (kt + 1) * 1024, b1);
      acc = __builtin_amdgcn_wmma_f32_16x16x32_bf16(false, a0.v, false, b0.v,
                                                    (short)0, acc, false, false);
      a0 = a1; b0 = b1;
    }
    acc = __builtin_amdgcn_wmma_f32_16x16x32_bf16(false, a0.v, false, b0.v,
                                                  (short)0, acc, false, false);
  }

  // --- pass 1: recurrent h contribution, 32 tiles of K=32 ---
  {
    const char* pa = (const char*)A1 + (size_t)mt * 32 * 1024 + laneByte;
    const char* pb = (const char*)B1 + (size_t)ntile * 32 * 1024 + laneByte;
    FragU a0, b0, a1, b1;
    load_frag_at(pa, 0, a0);
    load_frag_at(pb, 0, b0);
    #pragma unroll 4
    for (int kt = 0; kt < 31; ++kt) {
      load_frag_at(pa, (kt + 1) * 1024, a1);
      load_frag_at(pb, (kt + 1) * 1024, b1);
      acc = __builtin_amdgcn_wmma_f32_16x16x32_bf16(false, a0.v, false, b0.v,
                                                    (short)0, acc, false, false);
      a0 = a1; b0 = b1;
    }
    acc = __builtin_amdgcn_wmma_f32_16x16x32_bf16(false, a0.v, false, b0.v,
                                                  (short)0, acc, false, false);
  }

  const float bv = bias[g * 1024 + colInGate + (lane & 15)];
  const int rbase = (lane >> 4) << 3;  // C layout: VGPR r -> row r (lanes<16) / r+8
  #pragma unroll
  for (int r = 0; r < 8; ++r)
    lds[g][r + rbase][sub * 16 + (lane & 15)] = acc[r] + bv;
  __syncthreads();

  #pragma unroll
  for (int pass = 0; pass < 2; ++pass) {
    int idx  = tid + pass * 256;
    int row  = idx >> 5;
    int colL = idx & 31;
    int b_   = mt * 16 + row;
    int hcol = cg * 32 + colL;
    float iv = lds[0][row][colL];
    float fv = lds[1][row][colL];
    float gv = lds[2][row][colL];
    float ov = lds[3][row][colL];
    float cprev = cbuf[(size_t)b_ * 1024 + hcol];
    float si = 1.f / (1.f + __expf(-iv));
    float sf = 1.f / (1.f + __expf(-fv));
    float so = 1.f / (1.f + __expf(-ov));
    float cn = sf * cprev + si * tanhf(gv);
    float hn = so * tanhf(cn);
    cbuf[(size_t)b_ * 1024 + hcol] = cn;
    // write h into packed A-fragment layout (tile = (b>>4)*32 + cg)
    int kk   = hcol & 31;
    int half = (kk >> 3) & 1;
    int e    = (kk & 7) + ((kk >> 4) << 3);
    int l2   = (b_ & 15) + (half << 4);
    h_packed[(size_t)(((b_ >> 4) * 32 + (hcol >> 5)) * 512) + l2 * 16 + e] =
        f32_to_bf16_bits(hn);
    if (h_f32) h_f32[(size_t)b_ * 1024 + hcol] = hn;
  }
}

// ---------------- final projection ----------------
__global__ void final_proj(const float* __restrict__ h2, const float* __restrict__ Wfc,
                           const float* __restrict__ bfc, float* __restrict__ out) {
  int id = blockIdx.x * blockDim.x + threadIdx.x;
  if (id >= 128 * 60) return;
  int b = id / 60, c = id % 60;
  float s = bfc[c];
  const float* hr = h2 + (size_t)b * 1024;
  const float* wr = Wfc + (size_t)c * 1024;
  for (int k = 0; k < 1024; ++k) s += hr[k] * wr[k];
  out[(size_t)b * 60 + c] = s;
}

// ---------------- host launcher ----------------
extern "C" void kernel_launch(void* const* d_in, const int* in_sizes, int n_in,
                              void* d_out, int out_size, void* d_ws, size_t ws_size,
                              hipStream_t stream) {
  const float* x     = (const float*)d_in[0];
  const float* h1_0  = (const float*)d_in[1];
  const float* c1_0  = (const float*)d_in[2];
  const float* h2_0  = (const float*)d_in[3];
  const float* c2_0  = (const float*)d_in[4];
  const float* Wih1  = (const float*)d_in[5];
  const float* Whh1  = (const float*)d_in[6];
  const float* bih1  = (const float*)d_in[7];
  const float* bhh1  = (const float*)d_in[8];
  const float* Wih2  = (const float*)d_in[9];
  const float* Whh2  = (const float*)d_in[10];
  const float* bih2  = (const float*)d_in[11];
  const float* bhh2  = (const float*)d_in[12];
  const float* Wfc   = (const float*)d_in[13];
  const float* bfc   = (const float*)d_in[14];

  char* ws = (char*)d_ws;
  size_t off = 0;
  auto alloc = [&](size_t bytes) -> char* {
    char* p = ws + off;
    off += (bytes + 255) & ~(size_t)255;
    return p;
  };
  const size_t E_WIH1 = 256ull * 5 * 512;   // 655,360 packed bf16 elems
  const size_t E_WHH  = 256ull * 32 * 512;  // 4,194,304
  const size_t E_XP   = 256ull * 8 * 5 * 512;
  const size_t E_HP   = 8ull * 32 * 512;    // 131,072

  unsigned short* wih1p = (unsigned short*)alloc(E_WIH1 * 2);
  unsigned short* whh1p = (unsigned short*)alloc(E_WHH * 2);
  unsigned short* wih2p = (unsigned short*)alloc(E_WHH * 2);
  unsigned short* whh2p = (unsigned short*)alloc(E_WHH * 2);
  unsigned short* xp    = (unsigned short*)alloc(E_XP * 2);
  unsigned short* h1pA  = (unsigned short*)alloc(E_HP * 2);
  unsigned short* h1pB  = (unsigned short*)alloc(E_HP * 2);
  unsigned short* h2pA  = (unsigned short*)alloc(E_HP * 2);
  unsigned short* h2pB  = (unsigned short*)alloc(E_HP * 2);
  float* c1  = (float*)alloc(128ull * 1024 * 4);
  float* c2  = (float*)alloc(128ull * 1024 * 4);
  float* h2f = (float*)alloc(128ull * 1024 * 4);
  float* b1  = (float*)alloc(4096 * 4);
  float* b2  = (float*)alloc(4096 * 4);

  pack_weight_b<<<(int)((E_WIH1 + 255) / 256), 256, 0, stream>>>(Wih1, 150, 5, wih1p, (int)E_WIH1);
  pack_weight_b<<<(int)((E_WHH + 255) / 256), 256, 0, stream>>>(Whh1, 1024, 32, whh1p, (int)E_WHH);
  pack_weight_b<<<(int)((E_WHH + 255) / 256), 256, 0, stream>>>(Wih2, 1024, 32, wih2p, (int)E_WHH);
  pack_weight_b<<<(int)((E_WHH + 255) / 256), 256, 0, stream>>>(Whh2, 1024, 32, whh2p, (int)E_WHH);
  pack_x_a<<<(int)((E_XP + 255) / 256), 256, 0, stream>>>(x, xp, (int)E_XP);
  pack_h_a<<<(int)(E_HP / 256), 256, 0, stream>>>(h1_0, h1pA);
  pack_h_a<<<(int)(E_HP / 256), 256, 0, stream>>>(h2_0, h2pA);
  combine_bias<<<16, 256, 0, stream>>>(bih1, bhh1, bih2, bhh2, b1, b2);
  hipMemcpyAsync(c1, c1_0, 128ull * 1024 * 4, hipMemcpyDeviceToDevice, stream);
  hipMemcpyAsync(c2, c2_0, 128ull * 1024 * 4, hipMemcpyDeviceToDevice, stream);

  dim3 grid(8, 32);
  for (int t = 0; t < 256; ++t) {
    unsigned short* h1rd = (t & 1) ? h1pB : h1pA;
    unsigned short* h1wr = (t & 1) ? h1pA : h1pB;
    unsigned short* h2rd = (t & 1) ? h2pB : h2pA;
    unsigned short* h2wr = (t & 1) ? h2pA : h2pB;
    lstm_layer_kernel<<<grid, 256, 0, stream>>>(
        xp + (size_t)t * 8 * 5 * 512, 5, wih1p, h1rd, whh1p, b1, c1, h1wr, nullptr);
    lstm_layer_kernel<<<grid, 256, 0, stream>>>(
        h1wr, 32, wih2p, h2rd, whh2p, b2, c2, h2wr, h2f);
  }
  final_proj<<<30, 256, 0, stream>>>(h2f, Wfc, bfc, (float*)d_out);
}
